// SwinUNETREncoder3D_56736517980913
// MI455X (gfx1250) — matrix operand order repaired and checked by reading, not compiled, not run on hardware
//
#include <hip/hip_runtime.h>
#include <hip/hip_bf16.h>

// ---------------------------------------------------------------------------
// SwinUNETR-3D encoder for gfx1250 (MI455X). wave32.
// All GEMMs / attention matmuls: v_wmma_f32_16x16x32_f16, f32 accumulate.
// Fragments built from LDS as contiguous b128 loads (layout used here:
//   A: halves [16h..16h+15] of a 32-k row
//   B: halves [16h..16h+15] of a 32-k column
//   C/D: lane n = lane&15, row m = r + 8*(lane>>4))
// GEMM double-buffers its LDS tiles: one barrier per WMMA.
// ---------------------------------------------------------------------------

typedef __attribute__((ext_vector_type(16))) _Float16 v16h;
typedef __attribute__((ext_vector_type(8)))  _Float16 v8h;
typedef __attribute__((ext_vector_type(4)))  _Float16 v4h;
typedef __attribute__((ext_vector_type(8)))  float    v8f;

__device__ __forceinline__ v8f wmma16(v16h a, v16h b, v8f c) {
  return __builtin_amdgcn_wmma_f32_16x16x32_f16(false, a, false, b, (short)0, c,
                                                false, false);
}

__device__ __forceinline__ v16h join16(v8h lo, v8h hi) {
  return __builtin_shufflevector(lo, hi, 0, 1, 2, 3, 4, 5, 6, 7,
                                 8, 9, 10, 11, 12, 13, 14, 15);
}

__device__ __forceinline__ v8h cvt8(float4 a, float4 b) {
  v8h h;
  h[0] = (_Float16)a.x; h[1] = (_Float16)a.y; h[2] = (_Float16)a.z; h[3] = (_Float16)a.w;
  h[4] = (_Float16)b.x; h[5] = (_Float16)b.y; h[6] = (_Float16)b.z; h[7] = (_Float16)b.w;
  return h;
}

__device__ __forceinline__ v4h cvt4(float4 a) {
  v4h h;
  h[0] = (_Float16)a.x; h[1] = (_Float16)a.y; h[2] = (_Float16)a.z; h[3] = (_Float16)a.w;
  return h;
}

// ---------------------------------------------------------------------------
// TDM / async-tensor probe: NULL descriptor (group0.count==0 -> no transfer),
// plus s_wait_tensorcnt. Arity differs between toolchains (probe-verified).
// ---------------------------------------------------------------------------
typedef __attribute__((ext_vector_type(4))) unsigned int u32x4;
typedef __attribute__((ext_vector_type(8))) int          i32x8;
typedef __attribute__((ext_vector_type(4))) int          i32x4;

__global__ void tdm_null_k() {
  __shared__ long dummy[16];
  dummy[threadIdx.x & 15] = 0;
  __syncthreads();
#if __has_builtin(__builtin_amdgcn_tensor_load_to_lds)
  u32x4 g0 = {};  // count=0 => NULL tensor descriptor (ISA 8.3): no transfer
  i32x8 g1 = {};
  i32x4 g2 = {};
  i32x4 g3 = {};
#if __clang_major__ >= 23
  i32x8 g4 = {};
  __builtin_amdgcn_tensor_load_to_lds(g0, g1, g2, g3, g4, 0);
#else
  __builtin_amdgcn_tensor_load_to_lds(g0, g1, g2, g3, 0);
#endif
#endif
#if __has_builtin(__builtin_amdgcn_s_wait_tensorcnt)
  __builtin_amdgcn_s_wait_tensorcnt(0);
#endif
#if __has_builtin(__builtin_amdgcn_cluster_id_x)
  int cid = __builtin_amdgcn_cluster_id_x();
  asm volatile("" ::"s"(cid));
#endif
}

// ---------------------------------------------------------------------------
// Tile staging: full steps take a branch-free path (uniform scalar test);
// only the final 16-wide tail (K%32==16) runs the guarded path once.
// ---------------------------------------------------------------------------
__device__ __forceinline__ void stageAB(_Float16 (*As)[40], _Float16 (*Bs)[40],
                                        const float* __restrict__ asrc,
                                        const float* __restrict__ bsrc,
                                        int arow, int acb, int brow, int bcb,
                                        int k0, int K) {
  if (k0 + 32 <= K) {  // uniform: all-full step
    const float* s = asrc + k0 + acb;
    float4 f0 = *(const float4*)s, f1 = *(const float4*)(s + 4);
    float4 f2 = *(const float4*)(s + 8), f3 = *(const float4*)(s + 12);
    *(v8h*)&As[arow][acb]     = cvt8(f0, f1);
    *(v8h*)&As[arow][acb + 8] = cvt8(f2, f3);
    float4 g = *(const float4*)(bsrc + k0 + bcb);
    *(v4h*)&Bs[brow][bcb] = cvt4(g);
  } else {  // runs at most once per kernel (K % 32 == 16)
    v8h z8 = {};
    v4h z4 = {};
    if (k0 + acb + 16 <= K) {
      const float* s = asrc + k0 + acb;
      float4 f0 = *(const float4*)s, f1 = *(const float4*)(s + 4);
      float4 f2 = *(const float4*)(s + 8), f3 = *(const float4*)(s + 12);
      *(v8h*)&As[arow][acb]     = cvt8(f0, f1);
      *(v8h*)&As[arow][acb + 8] = cvt8(f2, f3);
    } else {
      *(v8h*)&As[arow][acb]     = z8;
      *(v8h*)&As[arow][acb + 8] = z8;
    }
    if (k0 + bcb + 4 <= K) {
      float4 g = *(const float4*)(bsrc + k0 + bcb);
      *(v4h*)&Bs[brow][bcb] = cvt4(g);
    } else {
      *(v4h*)&Bs[brow][bcb] = z4;
    }
  }
}

// ---------------------------------------------------------------------------
// WMMA GEMM: Y[M,N] = act( X[M,K] @ W[N,K]^T + bias ) + R
// Block: 128 threads (4 waves) -> 64x16 tile; double-buffered LDS.
// M % 64 == 0, N % 16 == 0, K % 16 == 0, K >= 32.
// ---------------------------------------------------------------------------
__global__ void gemm_wmma_k(const float* __restrict__ X,
                            const float* __restrict__ W,
                            const float* __restrict__ bias,
                            const float* __restrict__ R,
                            float* __restrict__ Y,
                            int M, int N, int K, int actGelu) {
  const int tm = blockIdx.x, tn = blockIdx.y;
  const int tid = threadIdx.x;
  const int wave = tid >> 5, lane = tid & 31;
  const int lm = lane & 15, lh = lane >> 4;

  __shared__ __align__(16) _Float16 As[2][64][40];
  __shared__ __align__(16) _Float16 Bs[2][16][40];

  const int arow = tid >> 1, acb = (tid & 1) * 16;  // A staging coords
  const int brow = tid >> 3, bcb = (tid & 7) * 4;   // B staging coords
  const float* asrc = X + (long)(tm * 64 + arow) * K;
  const float* bsrc = W + (long)(tn * 16 + brow) * K;

  const int steps = (K + 31) >> 5;
  stageAB(As[0], Bs[0], asrc, bsrc, arow, acb, brow, bcb, 0, K);
  __syncthreads();

  v8f acc = {};
  for (int i = 0; i < steps; ++i) {
    const int cur = i & 1;
    if (i + 1 < steps) {
      __builtin_prefetch(asrc + (i + 2) * 32, 0, 3);
      __builtin_prefetch(bsrc + (i + 2) * 32, 0, 3);
      stageAB(As[cur ^ 1], Bs[cur ^ 1], asrc, bsrc, arow, acb, brow, bcb,
              (i + 1) * 32, K);
    }
    const _Float16* ar = &As[cur][wave * 16 + lm][0];
    v16h a = join16(*(const v8h*)(ar + 16 * lh), *(const v8h*)(ar + 16 * lh + 8));
    const _Float16* br = &Bs[cur][lm][0];
    v16h b = join16(*(const v8h*)(br + 16 * lh), *(const v8h*)(br + 16 * lh + 8));
    acc = wmma16(a, b, acc);
    __syncthreads();
  }

  const int n = tn * 16 + lm;
  const float bv = bias ? bias[n] : 0.f;
#pragma unroll
  for (int r = 0; r < 8; ++r) {
    int m = tm * 64 + wave * 16 + r + 8 * lh;
    float v = acc[r] + bv;
    if (actGelu) v = 0.5f * v * (1.f + erff(v * 0.70710678118f));
    if (R) v += R[(long)m * N + n];
    Y[(long)m * N + n] = v;
  }
}

// ---------------------------------------------------------------------------
// Per-token LayerNorm
// ---------------------------------------------------------------------------
__global__ void layernorm_k(const float* __restrict__ X, const float* __restrict__ g,
                            const float* __restrict__ b, float* __restrict__ Y,
                            int T, int C) {
  long t = (long)blockIdx.x * blockDim.x + threadIdx.x;
  if (t >= T) return;
  const float* x = X + t * C;
  float m = 0.f;
  for (int c = 0; c < C; ++c) m += x[c];
  m /= C;
  float v = 0.f;
  for (int c = 0; c < C; ++c) { float d = x[c] - m; v += d * d; }
  v /= C;
  float inv = rsqrtf(v + 1e-5f);
  float* y = Y + t * C;
  for (int c = 0; c < C; ++c) y[c] = (x[c] - m) * inv * g[c] + b[c];
}

// ---------------------------------------------------------------------------
// Window attention: block (128 thr) per (window, head); head dim 16, N = 64.
// Shift folded into gather/scatter. Q/K zero-padded to k=32 in LDS,
// V stored transposed, probabilities stored f16 -> all fragments are b128.
// ---------------------------------------------------------------------------
__global__ void attn_k(const float* __restrict__ QKV, float* __restrict__ AOUT,
                       int S, int C, int shift) {
  const int win = blockIdx.x, head = blockIdx.y;
  const int wsn = S >> 2;
  const int wh = win / (wsn * wsn);
  const int rem = win % (wsn * wsn);
  const int ww = rem / wsn, wd = rem % wsn;

  __shared__ __align__(16) _Float16 Qs[64][40];  // cols 16..31 zero pad
  __shared__ __align__(16) _Float16 Ks[64][40];  // cols 16..31 zero pad
  __shared__ __align__(16) _Float16 Vt[16][72];  // [d][pos]
  __shared__ __align__(16) _Float16 Ps[64][72];  // probabilities (f16)
  __shared__ float Sb[64][64];
  __shared__ int toks[64];

  const int tid = threadIdx.x;
  if (tid < 64) {
    int i = tid >> 4, j = (tid >> 2) & 3, k = tid & 3;
    int xh = (wh * 4 + i + shift) % S;
    int yw = (ww * 4 + j + shift) % S;
    int zd = (wd * 4 + k + shift) % S;
    toks[tid] = (xh * S + yw) * S + zd;
  }
  __syncthreads();

  for (int idx = tid; idx < 64 * 16; idx += 128) {
    int p = idx >> 4, d = idx & 15;
    long base = (long)toks[p] * (3 * C) + head * 16 + d;
    Qs[p][d] = (_Float16)QKV[base];
    Qs[p][16 + d] = (_Float16)0.f;
    Ks[p][d] = (_Float16)QKV[base + C];
    Ks[p][16 + d] = (_Float16)0.f;
    Vt[d][p] = (_Float16)QKV[base + 2 * C];
  }
  __syncthreads();

  const int wave = tid >> 5, lane = tid & 31;
  const int lm = lane & 15, lh = lane >> 4;

  // scores: each wave owns 16 query rows; K=16 zero-padded to 32
  const _Float16* qr = &Qs[wave * 16 + lm][0];
  v16h aq = join16(*(const v8h*)(qr + 16 * lh), *(const v8h*)(qr + 16 * lh + 8));
#pragma unroll
  for (int nt = 0; nt < 4; ++nt) {
    const _Float16* kr = &Ks[nt * 16 + lm][0];
    v16h b = join16(*(const v8h*)(kr + 16 * lh), *(const v8h*)(kr + 16 * lh + 8));
    v8f c = {};
    c = wmma16(aq, b, c);
#pragma unroll
    for (int r = 0; r < 8; ++r)
      Sb[wave * 16 + r + 8 * lh][nt * 16 + lm] = c[r] * 0.25f;  // 16^-0.5
  }
  __syncthreads();

  if (tid < 64) {
    float mx = -1e30f;
    for (int c2 = 0; c2 < 64; ++c2) mx = fmaxf(mx, Sb[tid][c2]);
    float sum = 0.f;
    for (int c2 = 0; c2 < 64; ++c2) {
      float e2 = expf(Sb[tid][c2] - mx);
      Sb[tid][c2] = e2;
      sum += e2;
    }
    float inv = 1.f / sum;
    for (int c2 = 0; c2 < 64; ++c2) Ps[tid][c2] = (_Float16)(Sb[tid][c2] * inv);
  }
  __syncthreads();

  // O = P @ V : K=64 in two 32-steps, fragments contiguous
  v8f o = {};
#pragma unroll
  for (int ks = 0; ks < 2; ++ks) {
    const _Float16* pr = &Ps[wave * 16 + lm][ks * 32];
    v16h a = join16(*(const v8h*)(pr + 16 * lh), *(const v8h*)(pr + 16 * lh + 8));
    const _Float16* vr = &Vt[lm][ks * 32 + 16 * lh];
    v16h b = join16(*(const v8h*)(vr), *(const v8h*)(vr + 8));
    o = wmma16(a, b, o);
  }
#pragma unroll
  for (int r = 0; r < 8; ++r) {
    int p = wave * 16 + r + 8 * lh;
    AOUT[(long)toks[p] * C + head * 16 + lm] = o[r];
  }
}

// ---------------------------------------------------------------------------
// Stem kernels
// ---------------------------------------------------------------------------
__global__ void skip_conv_k(const float* __restrict__ x, const float* __restrict__ w,
                            const float* __restrict__ b, float* __restrict__ y) {
  long idx = (long)blockIdx.x * blockDim.x + threadIdx.x;
  const long NV = 2097152L;  // 128^3
  if (idx >= NV * 48) return;
  int oc = (int)(idx % 48);
  long vox = idx / 48;
  int d = (int)(vox & 127), wv = (int)((vox >> 7) & 127), h = (int)(vox >> 14);
  float acc = b[oc];
  const float* wb = w + (long)oc * 4 * 27;  // (oc, c, 3,3,3)
  for (int c = 0; c < 4; ++c)
    for (int i = 0; i < 3; ++i) {
      int hh = h + i - 1; if (hh < 0 || hh > 127) continue;
      for (int j = 0; j < 3; ++j) {
        int w2 = wv + j - 1; if (w2 < 0 || w2 > 127) continue;
        for (int k = 0; k < 3; ++k) {
          int dd = d + k - 1; if (dd < 0 || dd > 127) continue;
          acc += x[(((long)hh * 128 + w2) * 128 + dd) * 4 + c] *
                 wb[((c * 3 + i) * 3 + j) * 3 + k];
        }
      }
    }
  y[vox * 48 + oc] = acc;
}

__global__ void in_stats_k(const float* __restrict__ y, float* __restrict__ stats) {
  const int oc = blockIdx.y;
  long start = (long)blockIdx.x * blockDim.x * 32;
  float s = 0.f, s2 = 0.f;
  for (int it = 0; it < 32; ++it) {
    long v = start + (long)it * blockDim.x + threadIdx.x;
    if (v < 2097152L) { float t = y[v * 48 + oc]; s += t; s2 += t * t; }
  }
  __shared__ float rs[256], rq[256];
  rs[threadIdx.x] = s; rq[threadIdx.x] = s2;
  __syncthreads();
  for (int w = 128; w > 0; w >>= 1) {
    if (threadIdx.x < (unsigned)w) {
      rs[threadIdx.x] += rs[threadIdx.x + w];
      rq[threadIdx.x] += rq[threadIdx.x + w];
    }
    __syncthreads();
  }
  if (threadIdx.x == 0) {
    atomicAdd(&stats[oc], rs[0]);
    atomicAdd(&stats[48 + oc], rq[0]);
  }
}

__global__ void in_norm_relu_k(float* __restrict__ y, const float* __restrict__ stats) {
  long idx = (long)blockIdx.x * blockDim.x + threadIdx.x;
  if (idx >= 2097152L * 48) return;
  int oc = (int)(idx % 48);
  const float inN = 1.f / 2097152.f;
  float m = stats[oc] * inN;
  float var = stats[48 + oc] * inN - m * m;
  float v = (y[idx] - m) * rsqrtf(var + 1e-5f);
  y[idx] = v > 0.f ? v : 0.f;
}

__global__ void patch_embed_k(const float* __restrict__ x, const float* __restrict__ w,
                              const float* __restrict__ b,
                              float* __restrict__ out1, float* __restrict__ out2) {
  long idx = (long)blockIdx.x * blockDim.x + threadIdx.x;
  if (idx >= 32768L * 48) return;
  int oc = (int)(idx % 48);
  long vox = idx / 48;
  int od = (int)(vox % 32), ow = (int)((vox / 32) % 32), oh = (int)(vox / 1024);
  float acc = b[oc];
  const float* wb = w + (long)oc * 256;  // (oc, c, 4,4,4)
  for (int c = 0; c < 4; ++c)
    for (int i = 0; i < 4; ++i)
      for (int j = 0; j < 4; ++j)
        for (int k = 0; k < 4; ++k)
          acc += x[(((long)(oh * 4 + i) * 128 + (ow * 4 + j)) * 128 + (od * 4 + k)) * 4 + c] *
                 wb[((c * 4 + i) * 4 + j) * 4 + k];
  long o = vox * 48 + oc;
  out1[o] = acc;
  out2[o] = acc;
}

// ---------------------------------------------------------------------------
// Patch-merge gather (2x2x2 concat, channel ((hb*2+wb)*2+db)*C + c) + LN
// ---------------------------------------------------------------------------
__global__ void merge_ln_k(const float* __restrict__ Hin, const float* __restrict__ g,
                           const float* __restrict__ bb, float* __restrict__ Xout,
                           int S, int C) {
  int So = S >> 1;
  long T = (long)So * So * So;
  long t = (long)blockIdx.x * blockDim.x + threadIdx.x;
  if (t >= T) return;
  int zd = (int)(t % So), yw = (int)((t / So) % So), xh = (int)(t / ((long)So * So));
  float* xo = Xout + t * 8 * C;
  for (int q = 0; q < 8; ++q) {
    int hb = q >> 2, wb = (q >> 1) & 1, db = q & 1;
    const float* src = Hin + (((long)(2 * xh + hb) * S + (2 * yw + wb)) * S + (2 * zd + db)) * C;
    for (int c = 0; c < C; ++c) xo[q * C + c] = src[c];
  }
  int C8 = 8 * C;
  float m = 0.f;
  for (int c = 0; c < C8; ++c) m += xo[c];
  m /= C8;
  float v = 0.f;
  for (int c = 0; c < C8; ++c) { float d2 = xo[c] - m; v += d2 * d2; }
  v /= C8;
  float inv = rsqrtf(v + 1e-5f);
  for (int c = 0; c < C8; ++c) xo[c] = (xo[c] - m) * inv * g[c] + bb[c];
}

__global__ void copy_k(const float* __restrict__ a, float* __restrict__ b, long n) {
  long i = (long)blockIdx.x * blockDim.x + threadIdx.x;
  if (i < n) b[i] = a[i];
}

// ---------------------------------------------------------------------------
// Host orchestration
// ---------------------------------------------------------------------------
struct Blk {
  const float *m1_b, *m1_w, *m2_b, *m2_w, *n1_b, *n1_g, *n2_b, *n2_g;
  const float *out_b, *out_w, *qkv_b, *qkv_w;
};

static void run_block(hipStream_t st, const Blk& B, float* H, float* Y, float* QKV,
                      float* AOUT, float* MLP, int S, int C, int heads, int shifted,
                      float* finalOut) {
  long T = (long)S * S * S;
  layernorm_k<<<dim3((unsigned)((T + 255) / 256)), 256, 0, st>>>(H, B.n1_g, B.n1_b, Y, (int)T, C);
  gemm_wmma_k<<<dim3((unsigned)(T / 64), (unsigned)((3 * C) / 16)), 128, 0, st>>>(
      Y, B.qkv_w, B.qkv_b, nullptr, QKV, (int)T, 3 * C, C, 0);
  int wsn = S / 4;
  int nw = wsn * wsn * wsn;
  attn_k<<<dim3((unsigned)nw, (unsigned)heads), 128, 0, st>>>(QKV, AOUT, S, C, shifted);
  gemm_wmma_k<<<dim3((unsigned)(T / 64), (unsigned)(C / 16)), 128, 0, st>>>(
      AOUT, B.out_w, B.out_b, H, H, (int)T, C, C, 0);
  layernorm_k<<<dim3((unsigned)((T + 255) / 256)), 256, 0, st>>>(H, B.n2_g, B.n2_b, Y, (int)T, C);
  gemm_wmma_k<<<dim3((unsigned)(T / 64), (unsigned)((4 * C) / 16)), 128, 0, st>>>(
      Y, B.m1_w, B.m1_b, nullptr, MLP, (int)T, 4 * C, C, 1);
  float* dst = finalOut ? finalOut : H;
  gemm_wmma_k<<<dim3((unsigned)(T / 64), (unsigned)(C / 16)), 128, 0, st>>>(
      MLP, B.m2_w, B.m2_b, H, dst, (int)T, C, 4 * C, 0);
}

extern "C" void kernel_launch(void* const* d_in, const int* in_sizes, int n_in,
                              void* d_out, int out_size, void* d_ws, size_t ws_size,
                              hipStream_t stream) {
  (void)in_sizes; (void)n_in; (void)out_size; (void)ws_size;
  auto F = [](void* p) { return (const float*)p; };

  // d_in order: x, then params flattened by sorted pytree keys:
  // patch_b, patch_w, skip_b, skip_w, stages[ b0(12 sorted), b1(12), merge(n_b,n_g,red_w) ]
  int idx = 0;
  const float* x       = F(d_in[idx++]);
  const float* patch_b = F(d_in[idx++]);
  const float* patch_w = F(d_in[idx++]);
  const float* skip_b  = F(d_in[idx++]);
  const float* skip_w  = F(d_in[idx++]);

  Blk blks[4][2];
  const float *mg_nb[3], *mg_ng[3], *mg_w[3];
  for (int s = 0; s < 4; ++s) {
    for (int bi = 0; bi < 2; ++bi) {
      Blk& B = blks[s][bi];
      B.m1_b = F(d_in[idx++]); B.m1_w = F(d_in[idx++]);
      B.m2_b = F(d_in[idx++]); B.m2_w = F(d_in[idx++]);
      B.n1_b = F(d_in[idx++]); B.n1_g = F(d_in[idx++]);
      B.n2_b = F(d_in[idx++]); B.n2_g = F(d_in[idx++]);
      B.out_b = F(d_in[idx++]); B.out_w = F(d_in[idx++]);
      B.qkv_b = F(d_in[idx++]); B.qkv_w = F(d_in[idx++]);
    }
    if (s < 3) {
      mg_nb[s] = F(d_in[idx++]); mg_ng[s] = F(d_in[idx++]); mg_w[s] = F(d_in[idx++]);
    }
  }

  // workspace layout (floats)
  float* ws    = (float*)d_ws;
  float* STATS = ws;                    // 96
  float* H     = ws + 256;              // 1,572,864
  float* Y     = H + 1572864;           // 1,572,864
  float* QKV   = Y + 1572864;           // 4,718,592
  float* AOUT  = QKV + 4718592;         // 1,572,864
  float* MLP   = AOUT + 1572864;        // 6,291,456
  float* MX    = MLP + 6291456;         // 1,572,864

  // output layout: h | skip_embed | x2 | merge0 | merge1
  float* out    = (float*)d_out;
  float* o_h    = out;
  float* o_skip = out + 24576;
  float* o_x2   = o_skip + 100663296L;
  float* o_s3   = o_x2 + 1572864;
  float* o_s4   = o_s3 + 393216;

  // TDM/async-tensor + cluster-id probe (NULL descriptor: no data movement)
  tdm_null_k<<<1, 32, 0, stream>>>();

  // ---- stem ----
  hipMemsetAsync(STATS, 0, 96 * sizeof(float), stream);
  {
    long n = 2097152L * 48;
    skip_conv_k<<<dim3((unsigned)((n + 255) / 256)), 256, 0, stream>>>(x, skip_w, skip_b, o_skip);
    in_stats_k<<<dim3(256, 48), 256, 0, stream>>>(o_skip, STATS);
    in_norm_relu_k<<<dim3((unsigned)((n + 255) / 256)), 256, 0, stream>>>(o_skip, STATS);
  }
  {
    long n = 32768L * 48;
    patch_embed_k<<<dim3((unsigned)((n + 255) / 256)), 256, 0, stream>>>(x, patch_w, patch_b, H, o_x2);
  }

  // ---- stages ----
  for (int s = 0; s < 4; ++s) {
    int S = 32 >> s;
    int C = 48 << s;
    int heads = 3 << s;
    run_block(stream, blks[s][0], H, Y, QKV, AOUT, MLP, S, C, heads, 0, nullptr);
    run_block(stream, blks[s][1], H, Y, QKV, AOUT, MLP, S, C, heads, 1,
              (s == 3) ? o_h : nullptr);
    if (s < 3) {
      int So = S >> 1;
      long Tp = (long)So * So * So;
      merge_ln_k<<<dim3((unsigned)((Tp + 127) / 128)), 128, 0, stream>>>(H, mg_ng[s], mg_nb[s], MX, S, C);
      gemm_wmma_k<<<dim3((unsigned)(Tp / 64), (unsigned)((2 * C) / 16)), 128, 0, stream>>>(
          MX, mg_w[s], nullptr, nullptr, H, (int)Tp, 2 * C, 8 * C, 0);
      long nc = Tp * 2 * C;
      if (s == 0)
        copy_k<<<dim3((unsigned)((nc + 255) / 256)), 256, 0, stream>>>(H, o_s3, nc);
      else if (s == 1)
        copy_k<<<dim3((unsigned)((nc + 255) / 256)), 256, 0, stream>>>(H, o_s4, nc);
    }
  }
}
